// SpectralAttention_62397284876984
// MI455X (gfx1250) — compile-verified, matrix operand order checked by reading, not run on hardware
//
#include <hip/hip_runtime.h>

typedef __bf16 bf16;
typedef __attribute__((ext_vector_type(16))) __bf16 bf16x16;
typedef __attribute__((ext_vector_type(8))) float floatx8;

#define EPSV 1e-5f

static __device__ __forceinline__ floatx8 wmma_bf16(bf16x16 a, bf16x16 b, floatx8 c) {
  // v_wmma_f32_16x16x32_bf16: D = A(16x32) * B(32x16) + C
  return __builtin_amdgcn_wmma_f32_16x16x32_bf16(false, a, false, b, (short)0, c, false, false);
}

union Frag { bf16x16 v; uint4 q[2]; };

static __device__ __forceinline__ bf16x16 load_frag(const bf16* p) {
  Frag f;
  f.q[0] = *(const uint4*)(p);
  f.q[1] = *(const uint4*)(p + 16);
  return f.v;
}

// ---------------------------------------------------------------------------
// Generic NT GEMM: C[M,N] = A[M,K] (row-major bf16) * W[N,K]^T (row-major bf16)
// Block: 256 threads = 8 waves, each wave owns a 16(m) x 64(n) tile.
// Block tile: 128 x 64.  Optional batch over z (A offset + C column offset).
// ---------------------------------------------------------------------------
__global__ void gemm_nt_kernel(const bf16* __restrict__ A, long ldA, long batchStrideA,
                               const bf16* __restrict__ Bt, long ldB,
                               float* __restrict__ Cf, bf16* __restrict__ Cb,
                               long ldC, long batchColStride,
                               const float* __restrict__ bias,
                               int M, int N, int K)
{
  const int lane = threadIdx.x & 31;
  const int wave = threadIdx.x >> 5;
  const int m0 = blockIdx.y * 128 + wave * 16;
  const int n0 = blockIdx.x * 64;
  const long z = blockIdx.z;
  if (m0 >= M || n0 >= N) return;                 // wave-uniform
  const int lm = lane & 15;
  const int kb = (lane >> 4) * 8;                 // per-lane K sub-offset
  const bf16* Ab = A + z * batchStrideA;
  const long colBase = z * batchColStride + n0;

  floatx8 acc0 = {}, acc1 = {}, acc2 = {}, acc3 = {};
  const bf16* aRow  = Ab + (long)(m0 + lm) * ldA + kb;
  const bf16* bRow  = Bt + (long)(n0 + lm) * ldB + kb;

  for (int kk = 0; kk < K; kk += 32) {
    bf16x16 a  = load_frag(aRow + kk);
    bf16x16 b0 = load_frag(bRow + kk);
    bf16x16 b1 = load_frag(bRow + (long)16 * ldB + kk);
    bf16x16 b2 = load_frag(bRow + (long)32 * ldB + kk);
    bf16x16 b3 = load_frag(bRow + (long)48 * ldB + kk);
    acc0 = wmma_bf16(a, b0, acc0);
    acc1 = wmma_bf16(a, b1, acc1);
    acc2 = wmma_bf16(a, b2, acc2);
    acc3 = wmma_bf16(a, b3, acc3);
  }

  const int rOff = (lane >> 4) * 8;
  floatx8 accs[4] = {acc0, acc1, acc2, acc3};
#pragma unroll
  for (int nt = 0; nt < 4; ++nt) {
    const int n = n0 + nt * 16 + lm;
    const float bv = bias ? bias[n] : 0.f;
#pragma unroll
    for (int r = 0; r < 8; ++r) {
      const int m = m0 + rOff + r;
      const float val = accs[nt][r] + bv;
      const long idx = (long)m * ldC + colBase + nt * 16 + lm;
      if (Cb) Cb[idx] = (bf16)val;
      else    Cf[idx] = val;
    }
  }
}

// ---------------------------------------------------------------------------
// Build bf16 lower-triangular Toeplitz matrices T[k][t][s] = basis[t-s, k]
// ---------------------------------------------------------------------------
__global__ void build_T_kernel(const float* __restrict__ basis, bf16* __restrict__ T,
                               int L, int KF)
{
  const int t = blockIdx.x, kf = blockIdx.y;
  bf16* row = T + ((long)kf * L + t) * L;
  for (int s = threadIdx.x; s < L; s += blockDim.x) {
    float v = (s <= t) ? basis[(long)(t - s) * KF + kf] : 0.f;
    row[s] = (bf16)v;
  }
}

// Xt[d, s] = bf16(x[s, d])   (transposed input for the conv GEMM B operand)
__global__ void build_Xt_kernel(const float* __restrict__ x, bf16* __restrict__ Xt,
                                int L, int D)
{
  const int d = blockIdx.x;
  for (int s = threadIdx.x; s < L; s += blockDim.x)
    Xt[(long)d * L + s] = (bf16)x[(long)s * D + d];
}

__global__ void f32_to_bf16_kernel(const float* __restrict__ in, bf16* __restrict__ out, long n)
{
  for (long i = (long)blockIdx.x * blockDim.x + threadIdx.x; i < n;
       i += (long)gridDim.x * blockDim.x)
    out[i] = (bf16)in[i];
}

// ---------------------------------------------------------------------------
// Per-(token, head): L2-normalize k and v, sim = q.kn, qsum = sum(q),
// gate logit = kvs_h * vn^T Wg kn + b -> gz = lrelu2 + eps.
// Emits bf16 copies for the attention GEMMs (q row-major, vn row-major,
// knT transposed) and f32 vn for the scan. Block = 1 wave (32 lanes, 2 dims each).
// ---------------------------------------------------------------------------
__global__ void normgate_kernel(const float* __restrict__ q, const float* __restrict__ k,
                                const float* __restrict__ v,
                                const float* __restrict__ wg_w, const float* __restrict__ wg_b,
                                const float* __restrict__ kvs,
                                bf16* __restrict__ qB, bf16* __restrict__ vnB,
                                float* __restrict__ vnF, bf16* __restrict__ knT,
                                float* __restrict__ simA, float* __restrict__ gzA,
                                float* __restrict__ qsA, int L)
{
  const int t = blockIdx.x, h = blockIdx.y, l = threadIdx.x;
  const int d0 = l, d1 = l + 32;
  const long base = (long)t * 1024 + h * 64;

  float k0 = k[base + d0], k1 = k[base + d1];
  float v0 = v[base + d0], v1 = v[base + d1];
  float q0 = q[base + d0], q1 = q[base + d1];

  auto wred = [](float x) {
#pragma unroll
    for (int o = 16; o > 0; o >>= 1) x += __shfl_xor(x, o);
    return x;
  };

  float ksc = 1.f / fmaxf(sqrtf(wred(k0 * k0 + k1 * k1)), 1e-12f);
  float vsc = 1.f / fmaxf(sqrtf(wred(v0 * v0 + v1 * v1)), 1e-12f);
  float kn0 = k0 * ksc, kn1 = k1 * ksc;
  float vn0 = v0 * vsc, vn1 = v1 * vsc;
  float sim_t = wred(q0 * kn0 + q1 * kn1);
  float qs = wred(q0 + q1);

  __shared__ float knS[64];
  knS[d0] = kn0; knS[d1] = kn1;
  __syncthreads();
  float u0 = 0.f, u1 = 0.f;
#pragma unroll 4
  for (int n = 0; n < 64; ++n) {
    const float knn = knS[n];
    u0 += wg_w[d0 * 64 + n] * knn;
    u1 += wg_w[d1 * 64 + n] * knn;
  }
  float bil = wred(vn0 * u0 + vn1 * u1);
  float logit = kvs[h] * bil + wg_b[0];
  float y = logit > 0.f ? logit : 0.01f * logit;
  float g = y * y + EPSV;

  qB[base + d0] = (bf16)q0;  qB[base + d1] = (bf16)q1;
  vnB[base + d0] = (bf16)vn0; vnB[base + d1] = (bf16)vn1;
  vnF[base + d0] = vn0;       vnF[base + d1] = vn1;
  knT[(long)(h * 64 + d0) * L + t] = (bf16)kn0;
  knT[(long)(h * 64 + d1) * L + t] = (bf16)kn1;
  if (l == 0) { simA[h * L + t] = sim_t; gzA[h * L + t] = g; qsA[h * L + t] = qs; }
}

// ---------------------------------------------------------------------------
// Online-softmax prefix scan per head (sequential over t, lanes = head dims).
// ---------------------------------------------------------------------------
__global__ void scan_kernel(const float* __restrict__ simA, const float* __restrict__ gzA,
                            const float* __restrict__ vnF,
                            float* __restrict__ linA, float* __restrict__ wtA,
                            float* __restrict__ gcA, int L)
{
  const int h = blockIdx.x, d = threadIdx.x;   // 64 threads
  const float* simh = simA + h * L;
  const float* gzh  = gzA + h * L;
  float m = -INFINITY, nrm = 0.f, vacc = 0.f, gc = 0.f;
  for (int t = 0; t < L; ++t) {
    const float s = simh[t];
    const float mn = fmaxf(m, s);
    const float a = __expf(m - mn);
    const float e = __expf(s - mn);
    nrm = nrm * a + e;
    m = mn;
    vacc = vacc * a + e * vnF[(long)t * 1024 + h * 64 + d];
    gc += gzh[t];
    linA[(long)t * 1024 + h * 64 + d] = vacc / (nrm + EPSV);
    if (d == 0) { wtA[h * L + t] = e / (nrm + EPSV); gcA[h * L + t] = gc; }
  }
}

// ---------------------------------------------------------------------------
// Flash-style masked linear attention per head:
// Ybase_raw[t,n] = sum_{s<=t} gz[s] * (q[t].vn[s]) * kn[s,n]
// Block = 4 waves; wave w owns rows [t0, t0+15]; chained WMMAs with LDS
// relayout of the masked S tile (C-frag -> A-frag).
// ---------------------------------------------------------------------------
__global__ void attn_ybase_kernel(const bf16* __restrict__ qB, const bf16* __restrict__ vnB,
                                  const bf16* __restrict__ knT, const float* __restrict__ gzA,
                                  float* __restrict__ Ybase, int L)
{
  const int h = blockIdx.y;
  const int wave = threadIdx.x >> 5;
  const int lane = threadIdx.x & 31;
  const int t0 = blockIdx.x * 64 + wave * 16;
  const int lm = lane & 15;
  const int kb = (lane >> 4) * 8;
  const int rOff = (lane >> 4) * 8;

  __shared__ __align__(16) bf16 Sst[4][16][32];

  const bf16* qh  = qB + h * 64;
  const bf16* vh  = vnB + h * 64;
  const bf16* kTh = knT + (long)(h * 64) * L;
  const float* gzh = gzA + h * L;

  floatx8 y0 = {}, y1 = {}, y2 = {}, y3 = {};

  const int sbMax = (t0 + 15) / 32;
  for (int sb = 0; sb <= sbMax; ++sb) {
    const int s0 = sb * 32;
    floatx8 sacc0 = {}, sacc1 = {};
    // S tile (16 t x 32 s) = q . vn^T over d = 0..63
#pragma unroll
    for (int dk = 0; dk < 64; dk += 32) {
      bf16x16 a  = load_frag(qh + (long)(t0 + lm) * 1024 + dk + kb);
      bf16x16 b0 = load_frag(vh + (long)(s0 + lm) * 1024 + dk + kb);
      bf16x16 b1 = load_frag(vh + (long)(s0 + 16 + lm) * 1024 + dk + kb);
      sacc0 = wmma_bf16(a, b0, sacc0);
      sacc1 = wmma_bf16(a, b1, sacc1);
    }
    // causal mask + gz column scale; stage as bf16 in LDS for A-frag relayout
    const float gzl0 = gzh[s0 + lm];
    const float gzl1 = gzh[s0 + 16 + lm];
#pragma unroll
    for (int r = 0; r < 8; ++r) {
      const int t = t0 + rOff + r;
      const float m0 = (s0 + lm <= t) ? sacc0[r] * gzl0 : 0.f;
      const float m1 = (s0 + 16 + lm <= t) ? sacc1[r] * gzl1 : 0.f;
      Sst[wave][rOff + r][lm] = (bf16)m0;
      Sst[wave][rOff + r][16 + lm] = (bf16)m1;
    }
    asm volatile("s_wait_dscnt 0" ::: "memory");   // in-wave LDS RAW fence
    Frag fa;
    fa.q[0] = *(const uint4*)&Sst[wave][lm][kb];
    fa.q[1] = *(const uint4*)&Sst[wave][lm][kb + 16];
    bf16x16 a2 = fa.v;
    // Y += S' * kn  (B[k=s, n=d] from transposed knT, contiguous in s)
    bf16x16 bk0 = load_frag(kTh + (long)(0 * 16 + lm) * L + s0 + kb);
    bf16x16 bk1 = load_frag(kTh + (long)(1 * 16 + lm) * L + s0 + kb);
    bf16x16 bk2 = load_frag(kTh + (long)(2 * 16 + lm) * L + s0 + kb);
    bf16x16 bk3 = load_frag(kTh + (long)(3 * 16 + lm) * L + s0 + kb);
    y0 = wmma_bf16(a2, bk0, y0);
    y1 = wmma_bf16(a2, bk1, y1);
    y2 = wmma_bf16(a2, bk2, y2);
    y3 = wmma_bf16(a2, bk3, y3);
  }

  floatx8 ys[4] = {y0, y1, y2, y3};
#pragma unroll
  for (int nt = 0; nt < 4; ++nt)
#pragma unroll
    for (int r = 0; r < 8; ++r) {
      const int t = t0 + rOff + r;
      Ybase[(long)t * 1024 + h * 64 + nt * 16 + lm] = ys[nt][r];
    }
}

// ---------------------------------------------------------------------------
// Y = Ybase*kvs/(gc+eps) blended with Y_lin by softmax weight, then row L2-norm
// over all 1024 dims -> bf16 for the final output GEMM.
// ---------------------------------------------------------------------------
__global__ void combine_norm_kernel(const float* __restrict__ Ybase, const float* __restrict__ linA,
                                    const float* __restrict__ wtA, const float* __restrict__ gcA,
                                    const float* __restrict__ qsA, const float* __restrict__ kvs,
                                    bf16* __restrict__ Yb, int L)
{
  const int t = blockIdx.x, tid = threadIdx.x;
  float yv[4], ss = 0.f;
#pragma unroll
  for (int i = 0; i < 4; ++i) {
    const int j = tid + i * 256;
    const int h = j >> 6;
    const float base = Ybase[(long)t * 1024 + j] * kvs[h] / (gcA[h * L + t] + EPSV);
    const float li = qsA[h * L + t] * linA[(long)t * 1024 + j];
    const float w = wtA[h * L + t];
    const float y = base + (li - base) * w;
    yv[i] = y; ss += y * y;
  }
  __shared__ float red[256];
  red[tid] = ss;
  __syncthreads();
  for (int o = 128; o > 0; o >>= 1) {
    if (tid < o) red[tid] += red[tid + o];
    __syncthreads();
  }
  const float scale = 1.f / fmaxf(sqrtf(red[0]), 1e-12f);
#pragma unroll
  for (int i = 0; i < 4; ++i)
    Yb[(long)t * 1024 + tid + i * 256] = (bf16)(yv[i] * scale);
}

// ---------------------------------------------------------------------------
extern "C" void kernel_launch(void* const* d_in, const int* in_sizes, int n_in,
                              void* d_out, int out_size, void* d_ws, size_t ws_size,
                              hipStream_t stream)
{
  const int L = 1024, D = 1024, H = 16, KF = 16, KD = 16384;
  const float* x      = (const float*)d_in[0];
  const float* basis  = (const float*)d_in[1];
  const float* Mphi_w = (const float*)d_in[2];
  const float* Mphi_b = (const float*)d_in[3];
  const float* wq_w   = (const float*)d_in[4];
  const float* wq_b   = (const float*)d_in[5];
  const float* wk_w   = (const float*)d_in[6];
  const float* wk_b   = (const float*)d_in[7];
  const float* wv_w   = (const float*)d_in[8];
  const float* wv_b   = (const float*)d_in[9];
  const float* wo_w   = (const float*)d_in[10];
  const float* wo_b   = (const float*)d_in[11];
  const float* wg_w   = (const float*)d_in[12];
  const float* wg_b   = (const float*)d_in[13];
  const float* kvs    = (const float*)d_in[14];

  char* ws = (char*)d_ws;
  size_t off = 0;
  auto take = [&](size_t bytes) -> char* {
    char* p = ws + off;
    off = (off + bytes + 255) & ~(size_t)255;
    return p;
  };
  bf16* Tk    = (bf16*)take((size_t)KF * L * L * 2);   // Toeplitz matrices
  bf16* Xt    = (bf16*)take((size_t)D * L * 2);        // x transposed
  bf16* convf = (bf16*)take((size_t)L * KD * 2);       // conv output [L, K*D]
  bf16* MphiB = (bf16*)take((size_t)D * KD * 2);
  bf16* wqB   = (bf16*)take((size_t)D * D * 2);
  bf16* wkB   = (bf16*)take((size_t)D * D * 2);
  bf16* wvB   = (bf16*)take((size_t)D * D * 2);
  bf16* woB   = (bf16*)take((size_t)D * D * 2);
  bf16* xtB   = (bf16*)take((size_t)L * D * 2);        // x_tilde bf16
  float* qf   = (float*)take((size_t)L * D * 4);
  float* kf   = (float*)take((size_t)L * D * 4);
  float* vf   = (float*)take((size_t)L * D * 4);
  bf16* qB    = (bf16*)take((size_t)L * D * 2);
  bf16* vnB   = (bf16*)take((size_t)L * D * 2);
  bf16* knT   = (bf16*)take((size_t)D * L * 2);
  float* vnF  = (float*)take((size_t)L * D * 4);
  float* simA = (float*)take((size_t)H * L * 4);
  float* gzA  = (float*)take((size_t)H * L * 4);
  float* gcA  = (float*)take((size_t)H * L * 4);
  float* wtA  = (float*)take((size_t)H * L * 4);
  float* qsA  = (float*)take((size_t)H * L * 4);
  float* Ybase= (float*)take((size_t)L * D * 4);
  float* linA = (float*)take((size_t)L * D * 4);
  bf16* Yb    = (bf16*)take((size_t)L * D * 2);

  // --- Stage 0: prepare bf16 operands ---
  build_T_kernel<<<dim3(L, KF), 256, 0, stream>>>(basis, Tk, L, KF);
  build_Xt_kernel<<<dim3(D), 256, 0, stream>>>(x, Xt, L, D);
  f32_to_bf16_kernel<<<4096, 256, 0, stream>>>(Mphi_w, MphiB, (long)D * KD);
  f32_to_bf16_kernel<<<1024, 256, 0, stream>>>(wq_w, wqB, (long)D * D);
  f32_to_bf16_kernel<<<1024, 256, 0, stream>>>(wk_w, wkB, (long)D * D);
  f32_to_bf16_kernel<<<1024, 256, 0, stream>>>(wv_w, wvB, (long)D * D);
  f32_to_bf16_kernel<<<1024, 256, 0, stream>>>(wo_w, woB, (long)D * D);

  // --- Stage 1: causal spectral conv as 16 batched Toeplitz GEMMs ---
  gemm_nt_kernel<<<dim3(D / 64, L / 128, KF), 256, 0, stream>>>(
      Tk, L, (long)L * L, Xt, L, nullptr, convf, KD, D, nullptr, L, D, L);

  // --- Stage 2: M_phi projection [L,16384] x [16384,1024] ---
  gemm_nt_kernel<<<dim3(D / 64, L / 128, 1), 256, 0, stream>>>(
      convf, KD, 0, MphiB, KD, nullptr, xtB, D, 0, Mphi_b, L, D, KD);

  // --- Stage 3: q, k, v projections ---
  gemm_nt_kernel<<<dim3(D / 64, L / 128, 1), 256, 0, stream>>>(
      xtB, D, 0, wqB, D, qf, nullptr, D, 0, wq_b, L, D, D);
  gemm_nt_kernel<<<dim3(D / 64, L / 128, 1), 256, 0, stream>>>(
      xtB, D, 0, wkB, D, kf, nullptr, D, 0, wk_b, L, D, D);
  gemm_nt_kernel<<<dim3(D / 64, L / 128, 1), 256, 0, stream>>>(
      xtB, D, 0, wvB, D, vf, nullptr, D, 0, wv_b, L, D, D);

  // --- Stage 4: normalize / sim / gates, then the online-softmax scan ---
  normgate_kernel<<<dim3(L, H), 32, 0, stream>>>(qf, kf, vf, wg_w, wg_b, kvs,
                                                 qB, vnB, vnF, knT, simA, gzA, qsA, L);
  scan_kernel<<<dim3(H), 64, 0, stream>>>(simA, gzA, vnF, linA, wtA, gcA, L);

  // --- Stage 5: masked linear attention (Y_base) via chained WMMAs ---
  attn_ybase_kernel<<<dim3(L / 64, H), 128, 0, stream>>>(qB, vnB, knT, gzA, Ybase, L);

  // --- Stage 6: blend + L2 norm, final output projection ---
  combine_norm_kernel<<<dim3(L), 256, 0, stream>>>(Ybase, linA, wtA, gcA, qsA, kvs, Yb, L);
  gemm_nt_kernel<<<dim3(D / 64, L / 128, 1), 256, 0, stream>>>(
      Yb, D, 0, woB, D, (float*)d_out, nullptr, D, 0, wo_b, L, D, D);

  (void)in_sizes; (void)n_in; (void)out_size; (void)ws_size;
}
